// RWKV_TMix_x060_56203942035763
// MI455X (gfx1250) — compile-verified, matrix operand order checked by reading, not run on hardware
//
#include <hip/hip_runtime.h>

// ---------------- constants ----------------
#define CDIM   1024
#define TDIM   1024
#define BDIM   4
#define HHEAD  16
#define KHEAD  64
#define NROWS  (BDIM*TDIM)       // 4096
#define TMDIM  32
#define TDDIM  64
#define NCHUNK 16
#define CLEN   (TDIM/NCHUNK)     // 64
#define GN_EPS 6.4e-4f           // 1e-5 * 8^2

typedef __attribute__((ext_vector_type(16))) __bf16 v16bf;
typedef __attribute__((ext_vector_type(8)))  float  v8f;

__device__ __forceinline__ unsigned short bf16_rn(float f) {
    unsigned int u = __float_as_uint(f);
    u += 0x7fffu + ((u >> 16) & 1u);      // round-to-nearest-even
    return (unsigned short)(u >> 16);
}

// ---------------- 1) weight convert + transpose (f32 KxN -> bf16 NxK) ----------------
__global__ void __launch_bounds__(256)
w_to_bf16_t(const float* __restrict__ W, unsigned short* __restrict__ Wt) {
    int idx = blockIdx.x * 256 + threadIdx.x;       // idx = k*C + n
    int k = idx >> 10;
    int n = idx & (CDIM - 1);
    Wt[(size_t)n * CDIM + k] = bf16_rn(W[idx]);
}

// ---------------- 2) token-shift + LoRA mix + decay path ----------------
__global__ void __launch_bounds__(256)
prep_kernel(const float* __restrict__ x,
            const float* __restrict__ maa_x, const float* __restrict__ maa_w,
            const float* __restrict__ maa_k, const float* __restrict__ maa_v,
            const float* __restrict__ maa_r, const float* __restrict__ maa_g,
            const float* __restrict__ w1,    // (C, 5*TM)
            const float* __restrict__ w2,    // (5, TM, C)
            const float* __restrict__ tdecay,
            const float* __restrict__ tdw1,  // (C, TD)
            const float* __restrict__ tdw2,  // (TD, C)
            unsigned short* __restrict__ xk, unsigned short* __restrict__ xv,
            unsigned short* __restrict__ xr, unsigned short* __restrict__ xg,
            float* __restrict__ wdec) {
    const int row = blockIdx.x;            // b*T + t
    const int t   = row & (TDIM - 1);
    const int tid = threadIdx.x;

    __shared__ float xL[CDIM], xxL[CDIM], xxxL[CDIM], xwL[CDIM];
    __shared__ float m5[5 * TMDIM];
    __shared__ float tdL[TDDIM];

    const float* xrow = x + (size_t)row * CDIM;
    for (int c = tid; c < CDIM; c += 256) {
        float xc = xrow[c];
        float xs = (t == 0) ? 0.f : xrow[c - CDIM];
        float xxc = xs - xc;
        xL[c] = xc; xxL[c] = xxc;
        xxxL[c] = xc + xxc * maa_x[c];
    }
    __syncthreads();

    if (tid < 5 * TMDIM) {                 // 160 dots of length C (coalesced across lanes)
        float s = 0.f;
        for (int c = 0; c < CDIM; ++c) s += xxxL[c] * w1[c * (5 * TMDIM) + tid];
        m5[tid] = tanhf(s);
    }
    __syncthreads();

    for (int c = tid; c < CDIM; c += 256) {
        float mm[5];
#pragma unroll
        for (int f = 0; f < 5; ++f) {
            float s = 0.f;
            const float* w2f = w2 + (size_t)f * TMDIM * CDIM;
#pragma unroll
            for (int e = 0; e < TMDIM; ++e) s += m5[f * TMDIM + e] * w2f[(size_t)e * CDIM + c];
            mm[f] = s;
        }
        float xc = xL[c], xxc = xxL[c];
        xwL[c] = xc + xxc * (maa_w[c] + mm[0]);
        size_t o = (size_t)row * CDIM + c;
        xk[o] = bf16_rn(xc + xxc * (maa_k[c] + mm[1]));
        xv[o] = bf16_rn(xc + xxc * (maa_v[c] + mm[2]));
        xr[o] = bf16_rn(xc + xxc * (maa_r[c] + mm[3]));
        xg[o] = bf16_rn(xc + xxc * (maa_g[c] + mm[4]));
    }
    __syncthreads();

    if (tid < TDDIM) {
        float s = 0.f;
        for (int c = 0; c < CDIM; ++c) s += xwL[c] * tdw1[c * TDDIM + tid];
        tdL[tid] = tanhf(s);
    }
    __syncthreads();

    for (int c = tid; c < CDIM; c += 256) {
        float s = tdecay[c];
#pragma unroll
        for (int j = 0; j < TDDIM; ++j) s += tdL[j] * tdw2[(size_t)j * CDIM + c];
        wdec[(size_t)row * CDIM + c] = __expf(-__expf(s));   // decay in (0,1)
    }
}

// ---------------- 3) bf16 WMMA GEMM: C[M,N] = act(A[M,K] * Bt[N,K]^T) ----------------
// One wave -> 32(M) x 64(N): 8 accumulators, 2 A frags x 4 B frags per K-step
// (8 WMMA : 12 b128 loads). Block = 8 waves -> 128x128 tile.
__global__ void __launch_bounds__(256)
gemm_bf16_wmma(const unsigned short* __restrict__ A,   // M x K, row-major bf16
               const unsigned short* __restrict__ Bt,  // N x K, row-major bf16 (W transposed)
               float* __restrict__ C,
               int M, int N, int K, int act) {
    const int wave = threadIdx.x >> 5;
    const int lane = threadIdx.x & 31;
    const int l16  = lane & 15;
    const int hk   = (lane >> 4) << 3;          // 0 or 8: per-lane K sub-base
    const int mBase = blockIdx.x * 128 + (wave >> 1) * 32;
    const int nBase = blockIdx.y * 128 + (wave & 1) * 64;

    v8f acc[8];
#pragma unroll
    for (int i = 0; i < 8; ++i) acc[i] = v8f{};

    const unsigned short* aR0 = A  + (size_t)(mBase +  0 + l16) * K;
    const unsigned short* aR1 = A  + (size_t)(mBase + 16 + l16) * K;
    const unsigned short* bR0 = Bt + (size_t)(nBase +  0 + l16) * K;
    const unsigned short* bR1 = Bt + (size_t)(nBase + 16 + l16) * K;
    const unsigned short* bR2 = Bt + (size_t)(nBase + 32 + l16) * K;
    const unsigned short* bR3 = Bt + (size_t)(nBase + 48 + l16) * K;

    union Frag { v16bf v; uint4 q[2]; };

    for (int k0 = 0; k0 < K; k0 += 32) {
        Frag a0, a1, b0, b1, b2, b3;
        a0.q[0] = *(const uint4*)(aR0 + k0 + hk);
        a0.q[1] = *(const uint4*)(aR0 + k0 + 16 + hk);
        a1.q[0] = *(const uint4*)(aR1 + k0 + hk);
        a1.q[1] = *(const uint4*)(aR1 + k0 + 16 + hk);
        b0.q[0] = *(const uint4*)(bR0 + k0 + hk);
        b0.q[1] = *(const uint4*)(bR0 + k0 + 16 + hk);
        b1.q[0] = *(const uint4*)(bR1 + k0 + hk);
        b1.q[1] = *(const uint4*)(bR1 + k0 + 16 + hk);
        b2.q[0] = *(const uint4*)(bR2 + k0 + hk);
        b2.q[1] = *(const uint4*)(bR2 + k0 + 16 + hk);
        b3.q[0] = *(const uint4*)(bR3 + k0 + hk);
        b3.q[1] = *(const uint4*)(bR3 + k0 + 16 + hk);

        acc[0] = __builtin_amdgcn_wmma_f32_16x16x32_bf16(false, a0.v, false, b0.v, (short)0, acc[0], false, false);
        acc[1] = __builtin_amdgcn_wmma_f32_16x16x32_bf16(false, a0.v, false, b1.v, (short)0, acc[1], false, false);
        acc[2] = __builtin_amdgcn_wmma_f32_16x16x32_bf16(false, a0.v, false, b2.v, (short)0, acc[2], false, false);
        acc[3] = __builtin_amdgcn_wmma_f32_16x16x32_bf16(false, a0.v, false, b3.v, (short)0, acc[3], false, false);
        acc[4] = __builtin_amdgcn_wmma_f32_16x16x32_bf16(false, a1.v, false, b0.v, (short)0, acc[4], false, false);
        acc[5] = __builtin_amdgcn_wmma_f32_16x16x32_bf16(false, a1.v, false, b1.v, (short)0, acc[5], false, false);
        acc[6] = __builtin_amdgcn_wmma_f32_16x16x32_bf16(false, a1.v, false, b2.v, (short)0, acc[6], false, false);
        acc[7] = __builtin_amdgcn_wmma_f32_16x16x32_bf16(false, a1.v, false, b3.v, (short)0, acc[7], false, false);
    }

#pragma unroll
    for (int mt = 0; mt < 2; ++mt) {
        const int rBase = mBase + mt * 16 + ((lane >> 4) << 3);  // D: VGPR j -> row j + 8*(lane/16)
#pragma unroll
        for (int j = 0; j < 8; ++j) {
            size_t ro = (size_t)(rBase + j) * N + nBase + l16;
#pragma unroll
            for (int nt = 0; nt < 4; ++nt) {
                float o = acc[mt * 4 + nt][j];
                if (act) o = o / (1.f + __expf(-o));             // SiLU
                C[ro + nt * 16] = o;
            }
        }
    }
}

// ---------------- 4a) WKV pass 1: per-chunk local state (S_init = 0) + decay product ----
// S_t = wd_t (x)_k S_{t-1} + k_t v_t^T  is linear diagonal-decay => chunk-decomposable.
__global__ void __launch_bounds__(256)
wkv_pass1(const float* __restrict__ k, const float* __restrict__ v,
          const float* __restrict__ wd,
          float* __restrict__ Sloc, float* __restrict__ Pw) {
    const int bh = blockIdx.x;                    // b*H + h
    const int c  = blockIdx.y;                    // chunk
    const int b = bh >> 4, h = bh & (HHEAD - 1);
    const int tid = threadIdx.x;
    const int j  = tid & (KHEAD - 1);
    const int kg = tid >> 6;

    __shared__ float kL[KHEAD], vL[KHEAD], wL[KHEAD];

    float S[16];
#pragma unroll
    for (int i = 0; i < 16; ++i) S[i] = 0.f;
    float pw = 1.f;

    const size_t base = (size_t)b * TDIM * CDIM + h * KHEAD + (size_t)c * CLEN * CDIM;
    for (int t = 0; t < CLEN; ++t) {
        const size_t off = base + (size_t)t * CDIM;
        if      (tid < 64)  kL[tid]       = k [off + tid];
        else if (tid < 128) vL[tid - 64]  = v [off + tid - 64];
        else if (tid < 192) wL[tid - 128] = wd[off + tid - 128];
        __syncthreads();

        const float vj = vL[j];
#pragma unroll
        for (int i = 0; i < 16; ++i) {
            const int kk = kg * 16 + i;
            S[i] = S[i] * wL[kk] + kL[kk] * vj;
        }
        if (tid < 64) pw *= wL[tid];
        __syncthreads();
    }

    const size_t sb = ((size_t)bh * NCHUNK + c) * (KHEAD * KHEAD);
#pragma unroll
    for (int i = 0; i < 16; ++i)
        Sloc[sb + (size_t)(kg * 16 + i) * KHEAD + j] = S[i];
    if (tid < 64) Pw[((size_t)bh * NCHUNK + c) * KHEAD + tid] = pw;
}

// ---------------- 4b) WKV pass 2: prefix-combine chunk states (depth NCHUNK) ------------
__global__ void __launch_bounds__(256)
wkv_combine(const float* __restrict__ Sloc, const float* __restrict__ Pw,
            float* __restrict__ Sstart) {
    const int bh = blockIdx.x;
    const int tid = threadIdx.x;
    float cur[16];
#pragma unroll
    for (int i = 0; i < 16; ++i) cur[i] = 0.f;

    for (int c = 0; c < NCHUNK; ++c) {
        const size_t sb = ((size_t)bh * NCHUNK + c) * (KHEAD * KHEAD);
        const float* pwc = Pw + ((size_t)bh * NCHUNK + c) * KHEAD;
#pragma unroll
        for (int i = 0; i < 16; ++i) {
            const int e = tid + i * 256;          // element index k*64+j
            Sstart[sb + e] = cur[i];
            cur[i] = pwc[e >> 6] * cur[i] + Sloc[sb + e];
        }
    }
}

// ---------------- 4c) WKV pass 3: exact within-chunk scan from S_start, emit y ----------
__global__ void __launch_bounds__(256)
wkv_pass3(const float* __restrict__ r, const float* __restrict__ k,
          const float* __restrict__ v, const float* __restrict__ wd,
          const float* __restrict__ u, const float* __restrict__ Sstart,
          float* __restrict__ y) {
    const int bh = blockIdx.x;
    const int c  = blockIdx.y;
    const int b = bh >> 4, h = bh & (HHEAD - 1);
    const int tid = threadIdx.x;
    const int j  = tid & (KHEAD - 1);
    const int kg = tid >> 6;

    __shared__ float rL[KHEAD], kL[KHEAD], vL[KHEAD], wL[KHEAD], uL[KHEAD];
    __shared__ float yp[256];

    if (tid < KHEAD) uL[tid] = u[h * KHEAD + tid];

    const size_t sb = ((size_t)bh * NCHUNK + c) * (KHEAD * KHEAD);
    float S[16];
#pragma unroll
    for (int i = 0; i < 16; ++i)
        S[i] = Sstart[sb + (size_t)(kg * 16 + i) * KHEAD + j];

    const size_t base = (size_t)b * TDIM * CDIM + h * KHEAD + (size_t)c * CLEN * CDIM;
    for (int t = 0; t < CLEN; ++t) {
        const size_t off = base + (size_t)t * CDIM;
        if      (tid < 64)  rL[tid]       = r [off + tid];
        else if (tid < 128) kL[tid - 64]  = k [off + tid - 64];
        else if (tid < 192) vL[tid - 128] = v [off + tid - 128];
        else                wL[tid - 192] = wd[off + tid - 192];
        __syncthreads();

        const float vj = vL[j];
        float p = 0.f;
#pragma unroll
        for (int i = 0; i < 16; ++i) {
            const int kk = kg * 16 + i;
            const float kv = kL[kk] * vj;
            p += rL[kk] * (S[i] + uL[kk] * kv);   // y uses pre-update state
            S[i] = S[i] * wL[kk] + kv;
        }
        yp[tid] = p;
        __syncthreads();
        if (tid < 64)
            y[off + tid] = yp[tid] + yp[64 + tid] + yp[128 + tid] + yp[192 + tid];
    }
}

// ---------------- 5) per-head GroupNorm + gate -> bf16 A for output GEMM ----------------
__global__ void __launch_bounds__(64)
post_kernel(const float* __restrict__ y, const float* __restrict__ g,
            const float* __restrict__ gamma, const float* __restrict__ beta,
            unsigned short* __restrict__ afin) {
    const int rowh = blockIdx.x;                  // (b*T+t)*H + h
    const int h = rowh & (HHEAD - 1);
    const size_t base = (size_t)(rowh >> 4) * CDIM + h * KHEAD;
    const int tid = threadIdx.x;
    __shared__ float red[KHEAD];

    const float val = y[base + tid];
    red[tid] = val; __syncthreads();
#pragma unroll
    for (int s = 32; s > 0; s >>= 1) { if (tid < s) red[tid] += red[tid + s]; __syncthreads(); }
    const float mu = red[0] * (1.f / KHEAD);
    __syncthreads();

    const float d = val - mu;
    red[tid] = d * d; __syncthreads();
#pragma unroll
    for (int s = 32; s > 0; s >>= 1) { if (tid < s) red[tid] += red[tid + s]; __syncthreads(); }
    const float var = red[0] * (1.f / KHEAD);

    const float yn = d * rsqrtf(var + GN_EPS);
    const int cidx = h * KHEAD + tid;
    const float out = (yn * gamma[cidx] + beta[cidx]) * g[base + tid];
    afin[base + tid] = bf16_rn(out);
}

// ---------------- launch ----------------
extern "C" void kernel_launch(void* const* d_in, const int* in_sizes, int n_in,
                              void* d_out, int out_size, void* d_ws, size_t ws_size,
                              hipStream_t stream) {
    const float* x      = (const float*)d_in[0];
    const float* maa_x  = (const float*)d_in[1];
    const float* maa_w  = (const float*)d_in[2];
    const float* maa_k  = (const float*)d_in[3];
    const float* maa_v  = (const float*)d_in[4];
    const float* maa_r  = (const float*)d_in[5];
    const float* maa_g  = (const float*)d_in[6];
    const float* w1     = (const float*)d_in[7];
    const float* w2     = (const float*)d_in[8];
    const float* tdec   = (const float*)d_in[9];
    const float* tdw1   = (const float*)d_in[10];
    const float* tdw2   = (const float*)d_in[11];
    const float* faaaa  = (const float*)d_in[12];
    const float* W_r    = (const float*)d_in[13];
    const float* W_k    = (const float*)d_in[14];
    const float* W_v    = (const float*)d_in[15];
    const float* W_g    = (const float*)d_in[16];
    const float* W_o    = (const float*)d_in[17];
    const float* gamma  = (const float*)d_in[18];
    const float* beta   = (const float*)d_in[19];

    char* ws = (char*)d_ws;
    size_t off = 0;
    auto carve = [&](size_t bytes) { char* p = ws + off; off += (bytes + 255) & ~size_t(255); return p; };

    const size_t WB  = (size_t)CDIM * CDIM * sizeof(unsigned short);   // 2 MB
    const size_t AB  = (size_t)NROWS * CDIM * sizeof(unsigned short);  // 8 MB
    const size_t FB  = (size_t)NROWS * CDIM * sizeof(float);           // 16 MB
    const size_t SB  = (size_t)BDIM * HHEAD * NCHUNK * KHEAD * KHEAD * sizeof(float); // 16 MB
    const size_t PB  = (size_t)BDIM * HHEAD * NCHUNK * KHEAD * sizeof(float);         // 256 KB

    unsigned short* WtR = (unsigned short*)carve(WB);
    unsigned short* WtK = (unsigned short*)carve(WB);
    unsigned short* WtV = (unsigned short*)carve(WB);
    unsigned short* WtG = (unsigned short*)carve(WB);
    unsigned short* WtO = (unsigned short*)carve(WB);
    unsigned short* xk  = (unsigned short*)carve(AB);
    unsigned short* xv  = (unsigned short*)carve(AB);
    unsigned short* xr  = (unsigned short*)carve(AB);
    unsigned short* xg  = (unsigned short*)carve(AB);
    float* wdec   = (float*)carve(FB);
    float* rbuf   = (float*)carve(FB);
    float* kbuf   = (float*)carve(FB);
    float* vbuf   = (float*)carve(FB);
    float* gbuf   = (float*)carve(FB);
    float* ybuf   = (float*)carve(FB);
    float* Sloc   = (float*)carve(SB);
    float* Sstart = (float*)carve(SB);
    float* Pw     = (float*)carve(PB);
    unsigned short* afin = (unsigned short*)carve(AB);

    const int tgrid = (CDIM * CDIM) / 256;
    w_to_bf16_t<<<tgrid, 256, 0, stream>>>(W_r, WtR);
    w_to_bf16_t<<<tgrid, 256, 0, stream>>>(W_k, WtK);
    w_to_bf16_t<<<tgrid, 256, 0, stream>>>(W_v, WtV);
    w_to_bf16_t<<<tgrid, 256, 0, stream>>>(W_g, WtG);
    w_to_bf16_t<<<tgrid, 256, 0, stream>>>(W_o, WtO);

    prep_kernel<<<NROWS, 256, 0, stream>>>(x, maa_x, maa_w, maa_k, maa_v, maa_r, maa_g,
                                           w1, w2, tdec, tdw1, tdw2,
                                           xk, xv, xr, xg, wdec);

    dim3 ggrid(NROWS / 128, CDIM / 128);
    gemm_bf16_wmma<<<ggrid, 256, 0, stream>>>(xr, WtR, rbuf, NROWS, CDIM, CDIM, 0);
    gemm_bf16_wmma<<<ggrid, 256, 0, stream>>>(xk, WtK, kbuf, NROWS, CDIM, CDIM, 0);
    gemm_bf16_wmma<<<ggrid, 256, 0, stream>>>(xv, WtV, vbuf, NROWS, CDIM, CDIM, 0);
    gemm_bf16_wmma<<<ggrid, 256, 0, stream>>>(xg, WtG, gbuf, NROWS, CDIM, CDIM, 1);

    dim3 wgrid(BDIM * HHEAD, NCHUNK);
    wkv_pass1  <<<wgrid, 256, 0, stream>>>(kbuf, vbuf, wdec, Sloc, Pw);
    wkv_combine<<<BDIM * HHEAD, 256, 0, stream>>>(Sloc, Pw, Sstart);
    wkv_pass3  <<<wgrid, 256, 0, stream>>>(rbuf, kbuf, vbuf, wdec, faaaa, Sstart, ybuf);

    post_kernel<<<NROWS * HHEAD, 64, 0, stream>>>(ybuf, gbuf, gamma, beta, afin);

    gemm_bf16_wmma<<<ggrid, 256, 0, stream>>>(afin, WtO, (float*)d_out, NROWS, CDIM, CDIM, 0);
}
